// EnhancedGCNforOOD_88484916232739
// MI455X (gfx1250) — matrix operand order repaired and checked by reading, not compiled, not run on hardware
//
#include <hip/hip_runtime.h>
#include <hip/hip_bf16.h>
#include <math.h>

typedef __attribute__((ext_vector_type(16))) _Float16 v16h;
typedef __attribute__((ext_vector_type(8)))  _Float16 v8h;
typedef __attribute__((ext_vector_type(4)))  _Float16 v4h;
typedef __attribute__((ext_vector_type(8)))  float    v8f;
typedef __attribute__((ext_vector_type(4)))  float    v4f;

namespace {
constexpr int kN = 50000;
constexpr int kE = 600000;
constexpr int kF = 128;
constexpr int kH = 128;
constexpr int kG = 64;
constexpr int kP = 32;
constexpr int kC = 10;
constexpr int kHeads = 3;
constexpr int kK = 128;          // contraction dim of every WMMA GEMM here
constexpr int kKT = kK / 32;     // 4 K-tiles
constexpr float kBnEps  = 1e-5f;
constexpr float kCosEps = 1e-8f;
}

// ---------------- utility ----------------
__global__ void k_fill(float* p, float v, size_t n) {
  size_t i = (size_t)blockIdx.x * blockDim.x + threadIdx.x;
  size_t stride = (size_t)gridDim.x * blockDim.x;
  for (; i < n; i += stride) p[i] = v;
}

__device__ inline void atomicMaxF(float* addr, float val) {
  unsigned int* ua = (unsigned int*)addr;
  unsigned int old = *ua;
  while (__uint_as_float(old) < val) {
    unsigned int assumed = old;
    old = atomicCAS(ua, assumed, __float_as_uint(val));
    if (old == assumed) break;
  }
}

__device__ inline float eluf(float x) { return x > 0.f ? x : (expf(x) - 1.f); }

// ---------------- weight pack: W[128 x Ncols] -> B-fragment-ordered f16 ----------------
// Wp[((n*KT + kt)*32) + kr] = W[(kt*32 + kr)*Ncols + n], kr = half*16 + i
__global__ void k_pack_w(const float* __restrict__ W, _Float16* __restrict__ Wp, int Ncols) {
  int i = blockIdx.x * blockDim.x + threadIdx.x;
  if (i >= kK * Ncols) return;
  int k = i / Ncols, n = i % Ncols;
  int kt = k >> 5, kr = k & 31;
  Wp[((size_t)n * kKT + kt) * 32 + kr] = (_Float16)W[i];
}

// ---------------- WMMA GEMM helpers ----------------
__device__ inline void stage_A(const float* __restrict__ A, _Float16 (*As)[kK + 8],
                               int rowBase, int tid) {
#pragma unroll
  for (int it = 0; it < 2; ++it) {
    int i = tid + it * 256;            // 0..511, 32 float4 per row
    int r = i >> 5;
    int c4 = (i & 31) << 2;
    v4f a = *(const v4f*)&A[(size_t)(rowBase + r) * kK + c4];
    v4h h = { (_Float16)a[0], (_Float16)a[1], (_Float16)a[2], (_Float16)a[3] };
    *(v4h*)&As[r][c4] = h;             // ds_store_b64
  }
}

__device__ inline v8f mma_slab(const _Float16 (*As)[kK + 8], const _Float16* __restrict__ bbase,
                               int m, int half, v8f acc) {
#pragma unroll
  for (int kt = 0; kt < kKT; ++kt) {
    v8h a0 = *(const v8h*)&As[m][kt * 32 + half * 16];       // ds_load_b128
    v8h a1 = *(const v8h*)&As[m][kt * 32 + half * 16 + 8];
    v16h a = __builtin_shufflevector(a0, a1, 0,1,2,3,4,5,6,7,8,9,10,11,12,13,14,15);
    v8h b0 = *(const v8h*)(bbase + (size_t)kt * 32);         // global_load_b128 (L2 resident)
    v8h b1 = *(const v8h*)(bbase + (size_t)kt * 32 + 8);
    v16h b = __builtin_shufflevector(b0, b1, 0,1,2,3,4,5,6,7,8,9,10,11,12,13,14,15);
    acc = __builtin_amdgcn_wmma_f32_16x16x32_f16(false, a, false, b, (short)0, acc, false, false);
  }
  return acc;
}

// C[M,NCOLS] = A[M,128] @ B[128,NCOLS]; grid (M/16, NCOLS/128), block 256 (8 wave32s)
template <int NCOLS>
__global__ void k_gemm1(const float* __restrict__ A, const _Float16* __restrict__ Bp,
                        float* __restrict__ C) {
  __shared__ _Float16 As[16][kK + 8];  // +8 halfs pad -> conflict-free b128 row reads
  const int rowBase = blockIdx.x * 16;
  const int colBase = blockIdx.y * 128;
  const int tid = threadIdx.x, wave = tid >> 5, lane = tid & 31;
  const int m = lane & 15, half = lane >> 4;

  stage_A(A, As, rowBase, tid);
  __syncthreads();

  const int ncol = colBase + wave * 16 + m;
  const _Float16* bbase = Bp + (size_t)ncol * (kKT * 32) + half * 16;
  v8f acc = {};
  acc = mma_slab(As, bbase, m, half, acc);

  float* cp = C + (size_t)(rowBase + half * 8) * NCOLS + ncol;
#pragma unroll
  for (int r = 0; r < 8; ++r) cp[r * NCOLS] = acc[r];   // immediate-offset b32 stores
}

// C = elu(bn(A1@B1 + A2@B2 + bias)) ; K = 256 total -> 8 chained WMMAs, fused epilogue
template <int NCOLS>
__global__ void k_gemm2_bnelu(const float* __restrict__ A1, const _Float16* __restrict__ Bp1,
                              const float* __restrict__ A2, const _Float16* __restrict__ Bp2,
                              float* __restrict__ C, const float* __restrict__ bias,
                              const float* __restrict__ g, const float* __restrict__ be,
                              const float* __restrict__ m_, const float* __restrict__ v_) {
  __shared__ _Float16 As1[16][kK + 8];
  __shared__ _Float16 As2[16][kK + 8];
  const int rowBase = blockIdx.x * 16;
  const int colBase = blockIdx.y * 128;
  const int tid = threadIdx.x, wave = tid >> 5, lane = tid & 31;
  const int m = lane & 15, half = lane >> 4;

  stage_A(A1, As1, rowBase, tid);
  stage_A(A2, As2, rowBase, tid);
  __syncthreads();

  const int ncol = colBase + wave * 16 + m;
  const _Float16* b1 = Bp1 + (size_t)ncol * (kKT * 32) + half * 16;
  const _Float16* b2 = Bp2 + (size_t)ncol * (kKT * 32) + half * 16;
  v8f acc = {};
  acc = mma_slab(As1, b1, m, half, acc);
  acc = mma_slab(As2, b2, m, half, acc);

  // per-column epilogue params (uniform across the 8 rows this lane owns)
  float bi = bias[ncol];
  float scale = rsqrtf(v_[ncol] + kBnEps) * g[ncol];
  float mu = m_[ncol], bt = be[ncol];
  float* cp = C + (size_t)(rowBase + half * 8) * NCOLS + ncol;
#pragma unroll
  for (int r = 0; r < 8; ++r) {
    float t = (acc[r] + bi - mu) * scale + bt;
    cp[r * NCOLS] = eluf(t);
  }
}

// ---------------- graph kernels ----------------
__global__ void k_count(const int* __restrict__ dst, float* cnt) {
  int i = blockIdx.x * blockDim.x + threadIdx.x;
  if (i < kE) atomicAdd(&cnt[dst[i]], 1.0f);
}

__global__ void k_dinv(const float* __restrict__ cnt, float* dinv) {
  int i = blockIdx.x * blockDim.x + threadIdx.x;
  if (i < kN) dinv[i] = rsqrtf(cnt[i] + 1.0f);
}

// GCN: acc[dst] += h[src] * dinv[src]*dinv[dst]  (wave per edge, b128 gather)
__global__ void k_gcn_agg(const float* __restrict__ h, const int* __restrict__ src,
                          const int* __restrict__ dst, const float* __restrict__ dinv,
                          float* acc) {
  int e = blockIdx.x * (blockDim.x >> 5) + (threadIdx.x >> 5);
  if (e >= kE) return;
  int f0 = (threadIdx.x & 31) << 2;
  int s = src[e], d = dst[e];
  float coef = dinv[s] * dinv[d];
  v4f hv = *(const v4f*)(h + (size_t)s * kH + f0);
  float* ad = acc + (size_t)d * kH + f0;
#pragma unroll
  for (int j = 0; j < 4; ++j) atomicAdd(&ad[j], hv[j] * coef);
}

__global__ void k_h1_fin(float* acc, const float* __restrict__ h, const float* __restrict__ dinv,
                         const float* b1, const float* g, const float* be,
                         const float* m, const float* v) {
  size_t i = (size_t)blockIdx.x * blockDim.x + threadIdx.x;
  if (i >= (size_t)kN * kH) return;
  int n = (int)(i / kH), f = (int)(i % kH);
  float dv = dinv[n];
  float x = acc[i] + h[i] * dv * dv + b1[f];
  x = (x - m[f]) * rsqrtf(v[f] + kBnEps) * g[f] + be[f];
  acc[i] = eluf(x);
}

__global__ void k_sage_agg(const float* __restrict__ h1, const int* __restrict__ src,
                           const int* __restrict__ dst, float* acc) {
  int e = blockIdx.x * (blockDim.x >> 5) + (threadIdx.x >> 5);
  if (e >= kE) return;
  int f0 = (threadIdx.x & 31) << 2;
  int s = src[e], d = dst[e];
  v4f hv = *(const v4f*)(h1 + (size_t)s * kH + f0);
  float* ad = acc + (size_t)d * kH + f0;
#pragma unroll
  for (int j = 0; j < 4; ++j) atomicAdd(&ad[j], hv[j]);
}

__global__ void k_mean_fin(float* acc, const float* __restrict__ cnt) {
  size_t i = (size_t)blockIdx.x * blockDim.x + threadIdx.x;
  if (i >= (size_t)kN * kH) return;
  acc[i] /= fmaxf(cnt[i / kH], 1.0f);
}

// attention coefficients: al_src/al_dst [N*3]
__global__ void k_att_coef(const float* __restrict__ hg, const float* __restrict__ a_src,
                           const float* __restrict__ a_dst, float* als, float* ald) {
  int i = blockIdx.x * blockDim.x + threadIdx.x;
  if (i >= kN * kHeads) return;
  int n = i / kHeads, hd = i % kHeads;
  const float* hr = hg + (size_t)n * (kHeads * kH) + hd * kH;
  const float* as = a_src + hd * kH;
  const float* ad = a_dst + hd * kH;
  float s1 = 0.f, s2 = 0.f;
  for (int f = 0; f < kH; ++f) { float hv = hr[f]; s1 += hv * as[f]; s2 += hv * ad[f]; }
  als[i] = s1; ald[i] = s2;
}

__device__ inline void att_edge(int i, const int* src, const int* dst, int& s, int& d) {
  if (i < kE) { s = src[i]; d = dst[i]; } else { s = d = i - kE; }
}

__global__ void k_att_max(const int* __restrict__ src, const int* __restrict__ dst,
                          const float* __restrict__ als, const float* __restrict__ ald,
                          float* emax) {
  int i = blockIdx.x * blockDim.x + threadIdx.x;
  if (i >= kE + kN) return;
  int s, d; att_edge(i, src, dst, s, d);
#pragma unroll
  for (int hd = 0; hd < kHeads; ++hd) {
    float ev = als[s * kHeads + hd] + ald[d * kHeads + hd];
    ev = ev > 0.f ? ev : 0.2f * ev;
    atomicMaxF(&emax[d * kHeads + hd], ev);
  }
}

__global__ void k_att_den(const int* __restrict__ src, const int* __restrict__ dst,
                          const float* __restrict__ als, const float* __restrict__ ald,
                          const float* __restrict__ emax, float* den) {
  int i = blockIdx.x * blockDim.x + threadIdx.x;
  if (i >= kE + kN) return;
  int s, d; att_edge(i, src, dst, s, d);
#pragma unroll
  for (int hd = 0; hd < kHeads; ++hd) {
    float ev = als[s * kHeads + hd] + ald[d * kHeads + hd];
    ev = ev > 0.f ? ev : 0.2f * ev;
    atomicAdd(&den[d * kHeads + hd], expf(ev - emax[d * kHeads + hd]));
  }
}

__global__ void k_att_agg(const int* __restrict__ src, const int* __restrict__ dst,
                          const float* __restrict__ als, const float* __restrict__ ald,
                          const float* __restrict__ emax, const float* __restrict__ den,
                          const float* __restrict__ hg, float* h3acc) {
  int e = blockIdx.x * (blockDim.x >> 5) + (threadIdx.x >> 5);
  if (e >= kE + kN) return;
  int f0 = (threadIdx.x & 31) << 2;
  int s, d; att_edge(e, src, dst, s, d);
#pragma unroll
  for (int hd = 0; hd < kHeads; ++hd) {
    float ev = als[s * kHeads + hd] + ald[d * kHeads + hd];
    ev = ev > 0.f ? ev : 0.2f * ev;
    float alpha = expf(ev - emax[d * kHeads + hd]) / den[d * kHeads + hd];
    v4f hv = *(const v4f*)(hg + (size_t)s * (kHeads * kH) + hd * kH + f0);
    float* ar = h3acc + (size_t)d * (kHeads * kH) + hd * kH + f0;
#pragma unroll
    for (int j = 0; j < 4; ++j) atomicAdd(&ar[j], hv[j] * alpha);
  }
}

__global__ void k_h3_fin(const float* __restrict__ h3acc, const float* bg, const float* g,
                         const float* be, const float* m, const float* v, float* h3) {
  size_t i = (size_t)blockIdx.x * blockDim.x + threadIdx.x;
  if (i >= (size_t)kN * kH) return;
  int n = (int)(i / kH), f = (int)(i % kH);
  const float* a = h3acc + (size_t)n * (kHeads * kH);
  float x = (a[f] + a[kH + f] + a[2 * kH + f]) * (1.0f / 3.0f) + bg[f];
  x = (x - m[f]) * rsqrtf(v[f] + kBnEps) * g[f] + be[f];
  h3[i] = eluf(x);
}

// ---------------- pooling & heads ----------------
__global__ void k_pool_acc(const float* __restrict__ h3, const int* __restrict__ batch,
                           float* pooled, float* gcnt) {
  int n = blockIdx.x * (blockDim.x >> 5) + (threadIdx.x >> 5);
  if (n >= kN) return;
  int lane = threadIdx.x & 31;
  int f0 = lane << 2;
  int b = batch[n];
  v4f hv = *(const v4f*)(h3 + (size_t)n * kH + f0);
  float* pr = pooled + (size_t)b * kH + f0;
#pragma unroll
  for (int j = 0; j < 4; ++j) atomicAdd(&pr[j], hv[j]);
  if (lane == 0) atomicAdd(&gcnt[b], 1.0f);
}

__global__ void k_pool_fin(float* pooled, const float* __restrict__ gcnt) {
  int i = blockIdx.x * blockDim.x + threadIdx.x;
  if (i >= kG * kH) return;
  pooled[i] /= fmaxf(gcnt[i / kH], 1.0f);
}

__global__ void k_norm(const float* __restrict__ in, float* out, int rows) {
  int r = blockIdx.x * (blockDim.x >> 5) + (threadIdx.x >> 5);
  if (r >= rows) return;
  int lane = threadIdx.x & 31;
  const float* x = in + (size_t)r * kH;
  float ss = 0.f;
  for (int f = lane; f < kH; f += 32) { float t = x[f]; ss += t * t; }
#pragma unroll
  for (int off = 16; off; off >>= 1) ss += __shfl_xor(ss, off, 32);
  float inv = 1.0f / fmaxf(sqrtf(ss), kCosEps);
  float* o = out + (size_t)r * kH;
  for (int f = lane; f < kH; f += 32) o[f] = x[f] * inv;
}

// sims + softmax over P=32 protos: one wave per graph
__global__ void k_assign(const float* __restrict__ xn, const float* __restrict__ pn,
                         float* assign) {
  int g = blockIdx.x, p = threadIdx.x;
  const float* xr = xn + (size_t)g * kH;
  const float* pr = pn + (size_t)p * kH;
  float s = 0.f;
  for (int f = 0; f < kH; ++f) s += xr[f] * pr[f];
  float mx = s;
#pragma unroll
  for (int off = 16; off; off >>= 1) mx = fmaxf(mx, __shfl_xor(mx, off, 32));
  float ex = expf(s - mx);
  float sum = ex;
#pragma unroll
  for (int off = 16; off; off >>= 1) sum += __shfl_xor(sum, off, 32);
  assign[(size_t)g * kP + p] = ex / sum;
}

__global__ void k_proto_emb(const float* __restrict__ assign, const float* __restrict__ protos,
                            float* pe) {
  int g = blockIdx.x, h = threadIdx.x;
  float s = 0.f;
  for (int p = 0; p < kP; ++p) s += assign[g * kP + p] * protos[p * kH + h];
  pe[(size_t)g * kH + h] = s;
}

__global__ void k_logits(const float* __restrict__ pe, const float* __restrict__ Wc,
                         const float* __restrict__ bc, float* logits) {
  int g = blockIdx.x, c = threadIdx.x;
  if (c >= kC) return;
  float s = bc[c];
  for (int h = 0; h < kH; ++h) s += pe[g * kH + h] * Wc[h * kC + c];
  logits[g * kC + c] = s;
}

__global__ void k_dec1(const float* __restrict__ pooled, const float* __restrict__ Wd1,
                       const float* __restrict__ bd1, float* t1) {
  int g = blockIdx.x, j = threadIdx.x;  // 256 threads
  float s = bd1[j];
  for (int h = 0; h < kH; ++h) s += pooled[g * kH + h] * Wd1[h * (2 * kH) + j];
  t1[(size_t)g * (2 * kH) + j] = fmaxf(s, 0.f);
}

__global__ void k_dec2(const float* __restrict__ t1, const float* __restrict__ Wd2,
                       const float* __restrict__ bd2, float* rec) {
  int g = blockIdx.x, f = threadIdx.x;  // 128 threads
  float s = bd2[f];
  for (int j = 0; j < 2 * kH; ++j) s += t1[g * (2 * kH) + j] * Wd2[j * kF + f];
  rec[(size_t)g * kF + f] = s;
}

// ---------------- launch ----------------
extern "C" void kernel_launch(void* const* d_in, const int* in_sizes, int n_in,
                              void* d_out, int out_size, void* d_ws, size_t ws_size,
                              hipStream_t stream) {
  const float* x      = (const float*)d_in[0];
  const int*   ei     = (const int*)d_in[1];
  const int*   src    = ei;
  const int*   dst    = ei + kE;
  const int*   batch  = (const int*)d_in[2];
  const float* W1     = (const float*)d_in[3];
  const float* b1     = (const float*)d_in[4];
  const float* Wl     = (const float*)d_in[5];
  const float* Wr     = (const float*)d_in[6];
  const float* bs     = (const float*)d_in[7];
  const float* Wg     = (const float*)d_in[8];
  const float* a_src  = (const float*)d_in[9];
  const float* a_dst  = (const float*)d_in[10];
  const float* bg     = (const float*)d_in[11];
  const float* protos = (const float*)d_in[12];
  const float* Wc     = (const float*)d_in[13];
  const float* bc     = (const float*)d_in[14];
  const float* Wd1    = (const float*)d_in[15];
  const float* bd1    = (const float*)d_in[16];
  const float* Wd2    = (const float*)d_in[17];
  const float* bd2    = (const float*)d_in[18];
  const float* g1  = (const float*)d_in[19];
  const float* be1 = (const float*)d_in[20];
  const float* m1  = (const float*)d_in[21];
  const float* v1  = (const float*)d_in[22];
  const float* g2  = (const float*)d_in[23];
  const float* be2 = (const float*)d_in[24];
  const float* m2  = (const float*)d_in[25];
  const float* v2  = (const float*)d_in[26];
  const float* g3  = (const float*)d_in[27];
  const float* be3 = (const float*)d_in[28];
  const float* m3  = (const float*)d_in[29];
  const float* v3  = (const float*)d_in[30];

  float* ws = (float*)d_ws;
  size_t o = 0;
  float* CNT    = ws + o; o += kN;
  float* DINV   = ws + o; o += kN;
  float* HBUF   = ws + o; o += (size_t)kN * kH;          // h = x@W1 ; later reused as h3
  float* ACC1   = ws + o; o += (size_t)kN * kH;          // gcn acc -> h1 in place
  float* ACC2   = ws + o; o += (size_t)kN * kH;          // sage acc -> mean_nb in place
  float* H2     = ws + o; o += (size_t)kN * kH;
  float* HG     = ws + o; o += (size_t)kN * kHeads * kH;
  float* ALS    = ws + o; o += (size_t)kN * kHeads;
  float* ALD    = ws + o; o += (size_t)kN * kHeads;
  float* EMAX   = ws + o; o += (size_t)kN * kHeads;
  float* DEN    = ws + o; o += (size_t)kN * kHeads;
  float* H3ACC  = ws + o; o += (size_t)kN * kHeads * kH;
  float* POOLED = ws + o; o += (size_t)kG * kH;
  float* GCNT   = ws + o; o += kG;
  float* XN     = ws + o; o += (size_t)kG * kH;
  float* PN     = ws + o; o += (size_t)kP * kH;
  float* T1     = ws + o; o += (size_t)kG * 2 * kH;
  o = (o + 15) & ~(size_t)15;                            // 64B align packed-weight buffers
  _Float16* WP1 = (_Float16*)(ws + o); o += (size_t)kK * kH / 2;
  _Float16* WPl = (_Float16*)(ws + o); o += (size_t)kK * kH / 2;
  _Float16* WPr = (_Float16*)(ws + o); o += (size_t)kK * kH / 2;
  _Float16* WPg = (_Float16*)(ws + o); o += (size_t)kK * (kHeads * kH) / 2;
  float* H3     = HBUF;  // h no longer needed after h1 finalize

  float* out_logits = (float*)d_out;
  float* out_pe     = out_logits + kG * kC;
  float* out_assign = out_pe + kG * kH;
  float* out_rec    = out_assign + kG * kP;

  auto fill = [&](float* p, float v, size_t n) {
    unsigned blocks = (unsigned)((n + 255) / 256);
    if (blocks > 32768u) blocks = 32768u;
    k_fill<<<blocks, 256, 0, stream>>>(p, v, n);
  };
  fill(CNT, 0.f, kN);
  fill(ACC1, 0.f, (size_t)kN * kH);
  fill(ACC2, 0.f, (size_t)kN * kH);
  fill(EMAX, -INFINITY, (size_t)kN * kHeads);
  fill(DEN, 0.f, (size_t)kN * kHeads);
  fill(H3ACC, 0.f, (size_t)kN * kHeads * kH);
  fill(POOLED, 0.f, (size_t)kG * kH);
  fill(GCNT, 0.f, kG);

  // pack all weights once per launch (L2-resident, tiny)
  k_pack_w<<<(kK * kH + 255) / 256, 256, 0, stream>>>(W1, WP1, kH);
  k_pack_w<<<(kK * kH + 255) / 256, 256, 0, stream>>>(Wl, WPl, kH);
  k_pack_w<<<(kK * kH + 255) / 256, 256, 0, stream>>>(Wr, WPr, kH);
  k_pack_w<<<(kK * kHeads * kH + 255) / 256, 256, 0, stream>>>(Wg, WPg, kHeads * kH);

  const unsigned NH_BLOCKS = (unsigned)(((size_t)kN * kH + 255) / 256);
  const unsigned E_WAVE_BLOCKS = (kE + 7) / 8;
  const unsigned EN = kE + kN;

  // --- GCN ---
  k_count<<<(kE + 255) / 256, 256, 0, stream>>>(dst, CNT);
  k_dinv<<<(kN + 255) / 256, 256, 0, stream>>>(CNT, DINV);
  dim3 gH(kN / 16, 1);  // (3125,1)
  k_gemm1<kH><<<gH, 256, 0, stream>>>(x, WP1, HBUF);
  k_gcn_agg<<<E_WAVE_BLOCKS, 256, 0, stream>>>(HBUF, src, dst, DINV, ACC1);
  k_h1_fin<<<NH_BLOCKS, 256, 0, stream>>>(ACC1, HBUF, DINV, b1, g1, be1, m1, v1);

  // --- SAGE ---
  k_sage_agg<<<E_WAVE_BLOCKS, 256, 0, stream>>>(ACC1, src, dst, ACC2);
  k_mean_fin<<<NH_BLOCKS, 256, 0, stream>>>(ACC2, CNT);
  // h2 = elu(bn(mean_nb@Wl + h1@Wr + bs)) in one fused dual-GEMM (8 WMMAs)
  k_gemm2_bnelu<kH><<<gH, 256, 0, stream>>>(ACC2, WPl, ACC1, WPr, H2, bs, g2, be2, m2, v2);

  // --- GAT ---
  dim3 gG(kN / 16, (kHeads * kH) / 128);  // (3125,3)
  k_gemm1<kHeads * kH><<<gG, 256, 0, stream>>>(H2, WPg, HG);
  k_att_coef<<<(kN * kHeads + 255) / 256, 256, 0, stream>>>(HG, a_src, a_dst, ALS, ALD);
  k_att_max<<<(EN + 255) / 256, 256, 0, stream>>>(src, dst, ALS, ALD, EMAX);
  k_att_den<<<(EN + 255) / 256, 256, 0, stream>>>(src, dst, ALS, ALD, EMAX, DEN);
  k_att_agg<<<(EN + 7) / 8, 256, 0, stream>>>(src, dst, ALS, ALD, EMAX, DEN, HG, H3ACC);
  k_h3_fin<<<NH_BLOCKS, 256, 0, stream>>>(H3ACC, bg, g3, be3, m3, v3, H3);

  // --- pool + heads ---
  k_pool_acc<<<(kN + 7) / 8, 256, 0, stream>>>(H3, batch, POOLED, GCNT);
  k_pool_fin<<<(kG * kH + 255) / 256, 256, 0, stream>>>(POOLED, GCNT);
  k_norm<<<(kG + 7) / 8, 256, 0, stream>>>(POOLED, XN, kG);
  k_norm<<<(kP + 7) / 8, 256, 0, stream>>>(protos, PN, kP);
  k_assign<<<kG, 32, 0, stream>>>(XN, PN, out_assign);
  k_proto_emb<<<kG, kH, 0, stream>>>(out_assign, protos, out_pe);
  k_logits<<<kG, 32, 0, stream>>>(out_pe, Wc, bc, out_logits);
  k_dec1<<<kG, 2 * kH, 0, stream>>>(POOLED, Wd1, bd1, T1);
  k_dec2<<<kG, kF, 0, stream>>>(T1, Wd2, bd2, out_rec);
}
